// AttentionalGNN_69140383531597
// MI455X (gfx1250) — matrix operand order repaired and checked by reading, not compiled, not run on hardware
//
#include <hip/hip_runtime.h>
#include <hip/hip_bf16.h>

// ---------------------------------------------------------------------------
// AttentionalGNN (SuperGlue-style) for MI455X / gfx1250, wave32 + WMMA bf16.
// All contractions -> V_WMMA_F32_16X16X32_BF16 (fp32 accumulate).
// GEMM: 128x64 block tile, double-buffered LDS, 8 WMMA/wave/K-step.
// Message GEMM stages bf16 P via GLOBAL_LOAD_ASYNC_TO_LDS_B128 (ASYNCcnt).
// ---------------------------------------------------------------------------

#define USE_ASYNC_LDS 1

typedef __attribute__((ext_vector_type(16))) __bf16 v16bf;
typedef __attribute__((ext_vector_type(8)))  float  v8f;

static_assert(sizeof(__bf16) == 2, "bf16 must be 2 bytes");

constexpr int   L_  = 18;
constexpr int   D_  = 256;
constexpr int   H_  = 4;
constexpr int   B_  = 2;
constexpr int   N_  = 1024;
constexpr int   HD_ = 64;               // head dim
constexpr float SCALE_ = 0.125f;        // 1/sqrt(64)
constexpr float EPS_   = 1e-5f;

__device__ __forceinline__ __bf16 f2bf(float f) {
    union { float f; unsigned u; } v; v.f = f;
    unsigned r = v.u + 0x7FFFu + ((v.u >> 16) & 1u);   // round-to-nearest-even
    union { unsigned short s; __bf16 b; } o;
    o.s = (unsigned short)(r >> 16);
    return o.b;
}

#if USE_ASYNC_LDS
// Async global->LDS copy, 16 bytes/lane, tracked on ASYNCcnt (no VGPR data).
// VDST VGPR carries the LDS byte address; generic LDS pointers keep the LDS
// offset in their low bits, which is what the DGV unit consumes.
__device__ __forceinline__ void async_copy_b128(unsigned lds_addr, const void* gaddr) {
    asm volatile("global_load_async_to_lds_b128 %0, %1, off"
                 :: "v"(lds_addr), "v"((unsigned long long)(uintptr_t)gaddr)
                 : "memory");
}
__device__ __forceinline__ void wait_async0() {
    asm volatile("s_wait_asynccnt 0x0" ::: "memory");
}
#endif

// ---------------------------------------------------------------------------
// Generic GEMM:  Y[b, m, n] = (accum? Y : 0) + sum_k A[m, k] * X[b, k, n]
//                + bias[m] + res[b, m, n]
// A fp32 row-major with leading dim lda (allows column-sliced weight views).
// Block: 128 threads (4 waves). Block tile 128(M) x 64(N), K step 32,
// double-buffered LDS. Each wave: 32 rows x 64 cols -> 8 accumulators;
// each B fragment feeds two A fragments (8 WMMA per K-step per wave).
// Requires M % 128 == 0 (M is 256 or 512 here), N % 64 == 0, K % 32 == 0.
// ---------------------------------------------------------------------------
__global__ __launch_bounds__(128)
void gemm_wmma(const float* __restrict__ A, int lda,
               const float* __restrict__ X, long xBatch,
               const float* __restrict__ bias,
               const float* __restrict__ res, long resBatch,
               float* __restrict__ Y, long yBatch,
               int M, int N, int K, int accum)
{
    const int batch = blockIdx.z;
    const float* Xb = X + (long)batch * xBatch;
    float*       Yb = Y + (long)batch * yBatch;
    const int m0 = blockIdx.y * 128;
    const int n0 = blockIdx.x * 64;
    const int tid  = threadIdx.x;
    const int wave = tid >> 5;
    const int lane = tid & 31;

    __shared__ __bf16 As[2][128][32];   // [buf][m][k]
    __shared__ __bf16 Bs[2][64][32];    // [buf][n][k] (K-transposed)

    v8f acc[2][4] = {};

    auto stage = [&](int buf, int k0) {
        for (int idx = tid; idx < 128 * 32; idx += 128) {
            int r = idx >> 5, c = idx & 31;
            As[buf][r][c] = f2bf(A[(long)(m0 + r) * lda + (k0 + c)]);
        }
        for (int idx = tid; idx < 64 * 32; idx += 128) {
            int n = idx & 63, c = idx >> 6;
            Bs[buf][n][c] = f2bf(Xb[(long)(k0 + c) * N + (n0 + n)]);
        }
    };

    stage(0, 0);
    __syncthreads();

    for (int k0 = 0; k0 < K; k0 += 32) {
        const int cur = (k0 >> 5) & 1;
        if (k0 + 32 < K) {
            stage(cur ^ 1, k0 + 32);
            __builtin_prefetch(&A[(long)(m0 + tid) * lda + (k0 + 64)], 0, 1);
            __builtin_prefetch(&Xb[(long)(k0 + 64 + (tid >> 2)) * N + n0], 0, 1);
        }

        // Two A fragments (rows wave*32 .. wave*32+31), contiguous k = 16*half + i (matches B)
        const int koff = (lane >> 4) * 16;
        v16bf af[2];
        #pragma unroll
        for (int s = 0; s < 2; ++s) {
            const int arow = wave * 32 + s * 16 + (lane & 15);
            #pragma unroll
            for (int i = 0; i < 16; ++i) {
                af[s][i] = As[cur][arow][koff + i];
            }
        }
        const int bcol = lane & 15;
        const int kb   = (lane >> 4) * 16;
        #pragma unroll
        for (int t = 0; t < 4; ++t) {
            v16bf bf_;
            #pragma unroll
            for (int i = 0; i < 16; ++i) bf_[i] = Bs[cur][t * 16 + bcol][kb + i];
            acc[0][t] = __builtin_amdgcn_wmma_f32_16x16x32_bf16(
                false, af[0], false, bf_, (short)0, acc[0][t], false, false);
            acc[1][t] = __builtin_amdgcn_wmma_f32_16x16x32_bf16(
                false, af[1], false, bf_, (short)0, acc[1][t], false, false);
        }
        __syncthreads();
    }

    // epilogue: C/D layout -> lane half selects M+8, VGPR index = M row
    const int ccol = lane & 15;
    #pragma unroll
    for (int s = 0; s < 2; ++s) {
        const int crow = wave * 32 + s * 16 + ((lane >> 4) * 8);
        #pragma unroll
        for (int t = 0; t < 4; ++t) {
            #pragma unroll
            for (int r = 0; r < 8; ++r) {
                int m = m0 + crow + r;
                int n = n0 + t * 16 + ccol;
                long off = (long)m * N + n;
                float v = acc[s][t][r];
                if (bias) v += bias[m];
                if (res)  v += res[(long)batch * resBatch + off];
                if (accum) v += Yb[off];
                Yb[off] = v;
            }
        }
    }
}

// ---------------------------------------------------------------------------
// Attention scores: S[bh, n, m] = SCALE * sum_hd q[b, hd*H+h, n] * k[b, hd*H+h, m]
// grid (m_tiles=16, n_tiles=16, bh=8), block 128.
// ---------------------------------------------------------------------------
__global__ __launch_bounds__(128)
void attn_scores(const float* __restrict__ Q, const float* __restrict__ Kc,
                 float* __restrict__ S)
{
    const int bh = blockIdx.z, b = bh >> 2, h = bh & 3;
    const float* Qb = Q + (long)b * D_ * N_;
    const float* Kb = Kc + (long)b * D_ * N_;
    float*       Sb = S + (long)bh * N_ * N_;
    const int n0 = blockIdx.y * 64;
    const int m0 = blockIdx.x * 64;
    const int tid = threadIdx.x, wave = tid >> 5, lane = tid & 31;

    __shared__ __bf16 Aq[64][32];   // [n][hd]  (Q transposed on stage-in)
    __shared__ __bf16 Bk[64][32];   // [m][hd]

    v8f acc[4] = {};

    for (int k0 = 0; k0 < HD_; k0 += 32) {
        for (int idx = tid; idx < 64 * 32; idx += 128) {
            int r = idx & 63, c = idx >> 6;                // coalesced along n/m
            long ch = (long)((k0 + c) * H_ + h) * N_;      // head channel stride H
            Aq[r][c] = f2bf(Qb[ch + (n0 + r)]);
            Bk[r][c] = f2bf(Kb[ch + (m0 + r)]);
        }
        __syncthreads();

        const int arow = wave * 16 + (lane & 15);
        const int koff = (lane >> 4) * 16;
        v16bf af;
        #pragma unroll
        for (int i = 0; i < 16; ++i) {
            af[i] = Aq[arow][koff + i];
        }
        const int bcol = lane & 15;
        const int kb   = (lane >> 4) * 16;
        #pragma unroll
        for (int t = 0; t < 4; ++t) {
            v16bf bf_;
            #pragma unroll
            for (int i = 0; i < 16; ++i) bf_[i] = Bk[t * 16 + bcol][kb + i];
            acc[t] = __builtin_amdgcn_wmma_f32_16x16x32_bf16(
                false, af, false, bf_, (short)0, acc[t], false, false);
        }
        __syncthreads();
    }

    const int crow = wave * 16 + ((lane >> 4) * 8);
    const int ccol = lane & 15;
    #pragma unroll
    for (int t = 0; t < 4; ++t)
        #pragma unroll
        for (int r = 0; r < 8; ++r)
            Sb[(long)(n0 + crow + r) * N_ + (m0 + t * 16 + ccol)] = acc[t][r] * SCALE_;
}

// ---------------------------------------------------------------------------
// Row softmax over m; writes P directly in bf16 (feeds WMMA B-side of msg).
// grid (B*H*N = 8192 rows), block 256.
// ---------------------------------------------------------------------------
__global__ __launch_bounds__(256)
void softmax_rows(const float* __restrict__ S, __bf16* __restrict__ P)
{
    const long row = blockIdx.x;
    const float* Sr = S + row * N_;
    __bf16*      Pr = P + row * N_;
    const int tid = threadIdx.x;

    float v[4];
    float mx = -3.0e38f;
    #pragma unroll
    for (int i = 0; i < 4; ++i) { v[i] = Sr[tid + i * 256]; mx = fmaxf(mx, v[i]); }

    __shared__ float red[256];
    red[tid] = mx; __syncthreads();
    for (int s = 128; s > 0; s >>= 1) {
        if (tid < s) red[tid] = fmaxf(red[tid], red[tid + s]);
        __syncthreads();
    }
    mx = red[0]; __syncthreads();

    float sum = 0.f;
    #pragma unroll
    for (int i = 0; i < 4; ++i) { v[i] = __expf(v[i] - mx); sum += v[i]; }
    red[tid] = sum; __syncthreads();
    for (int s = 128; s > 0; s >>= 1) {
        if (tid < s) red[tid] += red[tid + s];
        __syncthreads();
    }
    const float inv = 1.0f / red[0];
    #pragma unroll
    for (int i = 0; i < 4; ++i) Pr[tid + i * 256] = f2bf(v[i] * inv);
}

// ---------------------------------------------------------------------------
// Message: msg[b, hd*H+h, n] = sum_m P[bh, n, m] * v[b, hd*H+h, m]
// GEMM view: M=HD(64 rows hd), N=1024(n), K=1024(m).  B-frag = P rows (contig,
// already bf16 -> staged with GLOBAL_LOAD_ASYNC_TO_LDS_B128 on ASYNCcnt).
// grid (n_tiles=16, 1, bh=8), block 128.
// ---------------------------------------------------------------------------
__global__ __launch_bounds__(128)
void attn_msg(const __bf16* __restrict__ P, const float* __restrict__ V,
              float* __restrict__ Mo)
{
    const int bh = blockIdx.z, b = bh >> 2, h = bh & 3;
    const __bf16* Pb = P + (long)bh * N_ * N_;
    const float*  Vb = V + (long)b * D_ * N_;
    float*        Mb = Mo + (long)b * D_ * N_;
    const int n0 = blockIdx.x * 64;
    const int tid = threadIdx.x, wave = tid >> 5, lane = tid & 31;

    __shared__ __bf16 Av[64][32];   // [hd][m]
    __shared__ __bf16 Bp[64][32];   // [n][m]

    v8f acc[4] = {};

    for (int k0 = 0; k0 < N_; k0 += 32) {
#if USE_ASYNC_LDS
        // P rows (bf16) -> LDS, 16B chunks: 64 rows x 4 chunks = 256 issues
        for (int idx = tid; idx < 256; idx += 128) {
            int r = idx >> 2, c8 = (idx & 3) * 8;
            async_copy_b128((unsigned)(uintptr_t)&Bp[r][c8],
                            &Pb[(long)(n0 + r) * N_ + (k0 + c8)]);
        }
#else
        for (int idx = tid; idx < 64 * 32; idx += 128) {
            int r = idx >> 5, c = idx & 31;
            Bp[r][c] = Pb[(long)(n0 + r) * N_ + (k0 + c)];
        }
#endif
        for (int idx = tid; idx < 64 * 32; idx += 128) {
            int r = idx >> 5, c = idx & 31;               // coalesced along m
            Av[r][c] = f2bf(Vb[(long)(r * H_ + h) * N_ + (k0 + c)]);
        }
        if (k0 + 32 < N_)
            __builtin_prefetch(&Pb[(long)(n0 + (tid >> 1)) * N_ + (k0 + 32)], 0, 1);
#if USE_ASYNC_LDS
        wait_async0();
#endif
        __syncthreads();

        const int arow = wave * 16 + (lane & 15);         // hd row
        const int koff = (lane >> 4) * 16;
        v16bf af;
        #pragma unroll
        for (int i = 0; i < 16; ++i) {
            af[i] = Av[arow][koff + i];
        }
        const int bcol = lane & 15;
        const int kb   = (lane >> 4) * 16;
        #pragma unroll
        for (int t = 0; t < 4; ++t) {
            v16bf bf_;
            #pragma unroll
            for (int i = 0; i < 16; ++i) bf_[i] = Bp[t * 16 + bcol][kb + i];
            acc[t] = __builtin_amdgcn_wmma_f32_16x16x32_bf16(
                false, af, false, bf_, (short)0, acc[t], false, false);
        }
        __syncthreads();
    }

    const int crow = wave * 16 + ((lane >> 4) * 8);
    const int ccol = lane & 15;
    #pragma unroll
    for (int t = 0; t < 4; ++t)
        #pragma unroll
        for (int r = 0; r < 8; ++r) {
            int hd = crow + r;
            int n  = n0 + t * 16 + ccol;
            Mb[(long)(hd * H_ + h) * N_ + n] = acc[t][r];
        }
}

// ---------------------------------------------------------------------------
// BatchNorm1d train-mode: biased stats per channel over (batch, length).
// ---------------------------------------------------------------------------
__global__ __launch_bounds__(256)
void bn_stats(const float* __restrict__ Hb, float* __restrict__ mu,
              float* __restrict__ rstd)
{
    const int c = blockIdx.x;                 // 0..511
    const int tid = threadIdx.x;
    float s = 0.f, ss = 0.f;
    for (int b = 0; b < B_; ++b)
        for (int n = tid; n < N_; n += 256) {
            float x = Hb[((long)b * (2 * D_) + c) * N_ + n];
            s += x; ss += x * x;
        }
    __shared__ float rs[256], rq[256];
    rs[tid] = s; rq[tid] = ss; __syncthreads();
    for (int k = 128; k > 0; k >>= 1) {
        if (tid < k) { rs[tid] += rs[tid + k]; rq[tid] += rq[tid + k]; }
        __syncthreads();
    }
    if (tid == 0) {
        const float inv = 1.0f / (float)(B_ * N_);
        float m   = rs[0] * inv;
        float var = rq[0] * inv - m * m;
        mu[c]   = m;
        rstd[c] = rsqrtf(var + EPS_);
    }
}

__global__ __launch_bounds__(256)
void bn_apply_relu(float* __restrict__ Hb, const float* __restrict__ mu,
                   const float* __restrict__ rstd, const float* __restrict__ g,
                   const float* __restrict__ be)
{
    long idx = (long)blockIdx.x * blockDim.x + threadIdx.x;
    if (idx >= (long)B_ * 2 * D_ * N_) return;
    int c = (int)((idx / N_) % (2 * D_));
    float x = Hb[idx];
    x = (x - mu[c]) * rstd[c] * g[c] + be[c];
    Hb[idx] = fmaxf(x, 0.0f);
}

// ---------------------------------------------------------------------------
// Host orchestration
// ---------------------------------------------------------------------------
extern "C" void kernel_launch(void* const* d_in, const int* in_sizes, int n_in,
                              void* d_out, int out_size, void* d_ws, size_t ws_size,
                              hipStream_t stream)
{
    (void)in_sizes; (void)n_in; (void)out_size; (void)ws_size;

    const float* desc0 = (const float*)d_in[0];
    const float* desc1 = (const float*)d_in[1];
    const float* Wq = (const float*)d_in[2];
    const float* bq = (const float*)d_in[3];
    const float* Wk = (const float*)d_in[4];
    const float* bk = (const float*)d_in[5];
    const float* Wv = (const float*)d_in[6];
    const float* bv = (const float*)d_in[7];
    const float* Wm = (const float*)d_in[8];
    const float* bm = (const float*)d_in[9];
    const float* W1 = (const float*)d_in[10];
    const float* b1 = (const float*)d_in[11];
    const float* g1 = (const float*)d_in[12];
    const float* be1 = (const float*)d_in[13];
    const float* W2 = (const float*)d_in[14];
    const float* b2 = (const float*)d_in[15];

    const long SZ  = (long)B_ * D_ * N_;          // 524288 floats per tensor
    const long HSZ = (long)B_ * 2 * D_ * N_;      // MLP hidden

    // workspace carve
    float* w = (float*)d_ws;
    float*  qc   = w; w += SZ;
    float*  kc   = w; w += SZ;
    float*  vc   = w; w += SZ;
    float*  msg  = w; w += SZ;
    float*  msg2 = w; w += SZ;
    float*  hbuf = w; w += HSZ;
    float*  mu   = w; w += 2 * D_;
    float*  rstd = w; w += 2 * D_;
    float*  Sbuf = w; w += (long)B_ * H_ * N_ * N_;
    __bf16* Pbuf = (__bf16*)w;                    // B*H*N*N bf16

    float* out = (float*)d_out;
    const dim3 blk128(128);

    auto gemm = [&](const float* A, int lda, const float* X, long xB,
                    const float* bias, const float* res, long resB,
                    float* Y, long yB, int M, int Ncols, int K, int accum) {
        dim3 grid(Ncols / 64, M / 128, B_);
        gemm_wmma<<<grid, blk128, 0, stream>>>(A, lda, X, xB, bias, res, resB,
                                               Y, yB, M, Ncols, K, accum);
    };

    auto prop = [&](int i, const float* x, const float* src,
                    const float* resid, float* outp) {
        const float* Wqi = Wq + (long)i * D_ * D_;
        const float* Wki = Wk + (long)i * D_ * D_;
        const float* Wvi = Wv + (long)i * D_ * D_;
        const float* Wmi = Wm + (long)i * D_ * D_;
        const float* W1i = W1 + (long)i * (2 * D_) * (2 * D_);
        const float* W2i = W2 + (long)i * D_ * (2 * D_);

        // Q/K/V projections
        gemm(Wqi, D_, x,   (long)D_ * N_, bq + (long)i * D_,
             nullptr, 0, qc, (long)D_ * N_, D_, N_, D_, 0);
        gemm(Wki, D_, src, (long)D_ * N_, bk + (long)i * D_,
             nullptr, 0, kc, (long)D_ * N_, D_, N_, D_, 0);
        gemm(Wvi, D_, src, (long)D_ * N_, bv + (long)i * D_,
             nullptr, 0, vc, (long)D_ * N_, D_, N_, D_, 0);

        // attention
        attn_scores<<<dim3(N_ / 64, N_ / 64, B_ * H_), blk128, 0, stream>>>(qc, kc, Sbuf);
        softmax_rows<<<dim3(B_ * H_ * N_), 256, 0, stream>>>(Sbuf, Pbuf);
        attn_msg<<<dim3(N_ / 64, 1, B_ * H_), blk128, 0, stream>>>(Pbuf, vc, msg);

        // merge conv
        gemm(Wmi, D_, msg, (long)D_ * N_, bm + (long)i * D_,
             nullptr, 0, msg2, (long)D_ * N_, D_, N_, D_, 0);

        // MLP layer 1 on concat([x, msg2]) via split-K (second call accumulates)
        gemm(W1i,      2 * D_, x,    (long)D_ * N_, b1 + (long)i * 2 * D_,
             nullptr, 0, hbuf, (long)2 * D_ * N_, 2 * D_, N_, D_, 0);
        gemm(W1i + D_, 2 * D_, msg2, (long)D_ * N_, nullptr,
             nullptr, 0, hbuf, (long)2 * D_ * N_, 2 * D_, N_, D_, 1);

        // BatchNorm (train-mode biased stats) + ReLU
        bn_stats<<<dim3(2 * D_), 256, 0, stream>>>(hbuf, mu, rstd);
        bn_apply_relu<<<dim3((int)((HSZ + 255) / 256)), 256, 0, stream>>>(
            hbuf, mu, rstd, g1 + (long)i * 2 * D_, be1 + (long)i * 2 * D_);

        // MLP layer 2 + residual fused
        gemm(W2i, 2 * D_, hbuf, (long)2 * D_ * N_, b2 + (long)i * D_,
             resid, (long)D_ * N_, outp, (long)D_ * N_, D_, N_, 2 * D_, 0);
    };

    // ---- layer 0 (self) ----
    prop(0, desc0, desc0, desc0, out + 0 * SZ);   // d_0
    prop(0, desc1, desc1, desc1, out + 1 * SZ);   // d_1

    // outs[2], outs[3] = original desc0/desc1
    hipMemcpyAsync(out + 2 * SZ, desc0, SZ * sizeof(float),
                   hipMemcpyDeviceToDevice, stream);
    hipMemcpyAsync(out + 3 * SZ, desc1, SZ * sizeof(float),
                   hipMemcpyDeviceToDevice, stream);

    // ---- layer 1 (cross), residual to ORIGINAL descriptors (reference quirk) ----
    const float* d0p = out + 0 * SZ;
    const float* d1p = out + 1 * SZ;
    prop(1, d0p, d1p, desc0, out + 4 * SZ);       // a
    prop(1, d1p, d0p, desc1, out + 5 * SZ);       // b

    const float* a  = out + 4 * SZ;
    const float* bb = out + 5 * SZ;
    for (int i = 2; i < L_; ++i) {
        const float* s0 = (i % 2 == 0) ? a  : bb;
        const float* s1 = (i % 2 == 0) ? bb : a;
        float* na = out + (long)(6 + (i - 2) * 2) * SZ;
        float* nb = na + SZ;
        prop(i, a,  s0, a,  na);
        prop(i, bb, s1, bb, nb);
        a = na; bb = nb;
    }
}
